// SelfAttention_8048768713377
// MI455X (gfx1250) — matrix operand order repaired and checked by reading, not compiled, not run on hardware
//
#include <hip/hip_runtime.h>
#include <cstdint>

typedef __bf16 bf16;
typedef __attribute__((ext_vector_type(16))) __bf16 v16bf;
typedef __attribute__((ext_vector_type(8)))  float  v8f;

#define D_DIM 1024
#define L_DIM 2048
#define B_DIM 4
#define MTOT  (B_DIM * L_DIM)
#define NEGV  (-1.0e9f)
#define LDSP  40   // padded LDS row stride (bf16): 80B -> rows 16B-aligned for b128 async writes

// ---------------- CDNA5 async global->LDS path (guarded; falls back to sync copies) ------------
#if defined(__has_builtin)
#  if __has_builtin(__builtin_amdgcn_global_load_async_to_lds_b128)
#    define HAVE_ASYNC_LDS 1
#  endif
#endif
#ifndef HAVE_ASYNC_LDS
#  define HAVE_ASYNC_LDS 0
#endif

#if HAVE_ASYNC_LDS
// Probe-confirmed param 0 type: 'int __attribute__((vector_size(16))) __device__ *'
typedef int v4i __attribute__((vector_size(16)));
typedef __attribute__((address_space(1))) v4i as1_v4i;
typedef __attribute__((address_space(3))) v4i as3_v4i;
__device__ __forceinline__ void async_copy16(const void* g, void* l) {
  // (global src v4i*, lds dst v4i*, imm offset, imm cpol); 16 bytes/lane, tracked by ASYNCcnt
  __builtin_amdgcn_global_load_async_to_lds_b128(
      (as1_v4i*)(uintptr_t)g, (as3_v4i*)(unsigned)(uintptr_t)l, 0, 0);
}
#endif

__device__ __forceinline__ void wait_async0() {
#if HAVE_ASYNC_LDS
#  if __has_builtin(__builtin_amdgcn_s_wait_asynccnt)
  __builtin_amdgcn_s_wait_asynccnt(0);
#  else
  asm volatile("s_wait_asynccnt 0" ::: "memory");
#  endif
#endif
}

union Frag16 { v16bf v; unsigned u[8]; };

__device__ __forceinline__ v8f wmma_bf16f32(v16bf a, v16bf b, v8f c) {
  // (neg_a, A, neg_b, B, c_mod, C, reuse_a, reuse_b)
  return __builtin_amdgcn_wmma_f32_16x16x32_bf16(false, a, false, b, (short)0, c, false, false);
}

// A fragment (16x32, M across lanes 0..15): pairs e=(2j,2j+1) -> K = 2j + 8h + (j>=4 ? 8 : 0)
__device__ __forceinline__ void load_afrag(const bf16 (*sT)[LDSP], int rbase, int lane, Frag16& f) {
  const int h = lane >> 4;
  const int m = rbase + (lane & 15);
#pragma unroll
  for (int j = 0; j < 8; ++j) {
    const int kk = 2 * j + 8 * h + ((j >= 4) ? 8 : 0);
    f.u[j] = *(const unsigned*)&sT[m][kk];
  }
}

// B fragment (32x16, N across lanes 0..15): pairs e=(2j,2j+1) -> K = 2j + 8h + (j>=4 ? 8 : 0)
__device__ __forceinline__ void load_bfrag(const bf16 (*sT)[LDSP], int nbase, int lane, Frag16& f) {
  const int h = lane >> 4;
  const int n = nbase + (lane & 15);
#pragma unroll
  for (int j = 0; j < 8; ++j) {
    const int kk = 2 * j + 8 * h + ((j >= 4) ? 8 : 0);
    f.u[j] = *(const unsigned*)&sT[n][kk];
  }
}

// One K=32 step of a 128x128 block tile. Wave (wr,wc) owns rows wr*32..+31, cols wc*64..+63.
__device__ __forceinline__ void mma_step(const bf16 (*sA)[LDSP], const bf16 (*sB)[LDSP],
                                         int lane, int wr, int wc, v8f acc[2][4]) {
  Frag16 a[2], b[4];
#pragma unroll
  for (int mi = 0; mi < 2; ++mi) load_afrag(sA, wr * 32 + mi * 16, lane, a[mi]);
#pragma unroll
  for (int ni = 0; ni < 4; ++ni) load_bfrag(sB, wc * 64 + ni * 16, lane, b[ni]);
#pragma unroll
  for (int mi = 0; mi < 2; ++mi)
#pragma unroll
    for (int ni = 0; ni < 4; ++ni)
      acc[mi][ni] = wmma_bf16f32(a[mi].v, b[ni].v, acc[mi][ni]);
}

// ---------------- Kernel 1: QKV projection (f32 in -> bf16 out), z selects q/k/v ----------------
// Software-pipelined: next chunk's global loads issued before this chunk's WMMAs.
__global__ __launch_bounds__(256)
void qkv_gemm_kernel(const float* __restrict__ x,
                     const float* __restrict__ Wq, const float* __restrict__ bq,
                     const float* __restrict__ Wk, const float* __restrict__ bk,
                     const float* __restrict__ Wv, const float* __restrict__ bv,
                     bf16* __restrict__ qo, bf16* __restrict__ ko, bf16* __restrict__ vTo)
{
  __shared__ __align__(16) bf16 sA[128][LDSP];   // x tile [m][k]
  __shared__ __align__(16) bf16 sB[128][LDSP];   // W tile transposed [n][k]
  const int which = blockIdx.z;
  const float* __restrict__ W    = (which == 0) ? Wq : (which == 1) ? Wk : Wv;
  const float* __restrict__ bias = (which == 0) ? bq : (which == 1) ? bk : bv;
  const int tid = threadIdx.x, lane = tid & 31, wave = tid >> 5;
  const int wr = wave & 3, wc = wave >> 2;
  const int m0 = blockIdx.x * 128, n0 = blockIdx.y * 128;

  v8f acc[2][4] = {};
  float rx[16], rw[16];

  auto gload = [&](int k0) {
#pragma unroll
    for (int it = 0; it < 16; ++it) {
      const int idx = tid + it * 256;
      rx[it] = x[(size_t)(m0 + (idx >> 5)) * D_DIM + k0 + (idx & 31)];
      rw[it] = W[(size_t)(k0 + (idx >> 7)) * D_DIM + n0 + (idx & 127)];
    }
  };

  gload(0);
  for (int k0 = 0; k0 < D_DIM; k0 += 32) {
#pragma unroll
    for (int it = 0; it < 16; ++it) {
      const int idx = tid + it * 256;
      sA[idx >> 5][idx & 31]  = (bf16)rx[it];
      sB[idx & 127][idx >> 7] = (bf16)rw[it];   // transpose W into [n][k]
    }
    __syncthreads();
    if (k0 + 32 < D_DIM) gload(k0 + 32);        // overlap next loads with WMMA
    mma_step(sA, sB, lane, wr, wc, acc);
    __syncthreads();
  }

  const int h = lane >> 4;
#pragma unroll
  for (int mi = 0; mi < 2; ++mi)
#pragma unroll
    for (int ni = 0; ni < 4; ++ni) {
      const int n = n0 + wc * 64 + ni * 16 + (lane & 15);
      const float bn = bias[n];
#pragma unroll
      for (int r = 0; r < 8; ++r) {
        const int m = m0 + wr * 32 + mi * 16 + r + 8 * h;
        const bf16 o = (bf16)(acc[mi][ni][r] + bn);
        if (which == 2) {                       // V stored transposed per batch: vT[b][d][j]
          const int bb = m >> 11;               // / L_DIM
          const int i  = m & (L_DIM - 1);
          vTo[((size_t)bb * D_DIM + n) * L_DIM + i] = o;
        } else if (which == 0) {
          qo[(size_t)m * D_DIM + n] = o;
        } else {
          ko[(size_t)m * D_DIM + n] = o;
        }
      }
    }
}

// ---------------- Kernel 2: S = scale * Q.K^T + causal + length masks (bf16 out) ----------------
// Double-buffered LDS with async global->LDS copies overlapping the WMMAs.
__global__ __launch_bounds__(256)
void scores_kernel(const bf16* __restrict__ qb, const bf16* __restrict__ kb,
                   const int* __restrict__ lengths, bf16* __restrict__ P)
{
  const int i0 = blockIdx.x * 128;
  const int j0 = blockIdx.y * 128;
  if (j0 > i0 + 127) return;                    // fully-masked tile: never read downstream
  const int b = blockIdx.z;
  const int len = lengths[b];

  __shared__ __align__(16) bf16 sQ[2][128][LDSP];
  __shared__ __align__(16) bf16 sK[2][128][LDSP];
  const int tid = threadIdx.x, lane = tid & 31, wave = tid >> 5;
  const int wr = wave & 3, wc = wave >> 2;
  const bf16* __restrict__ qrow = qb + (size_t)b * L_DIM * D_DIM;
  const bf16* __restrict__ krow = kb + (size_t)b * L_DIM * D_DIM;

  v8f acc[2][4] = {};

  auto stage = [&](int buf, int d0) {
#if HAVE_ASYNC_LDS
#pragma unroll
    for (int it = 0; it < 2; ++it) {            // 512 lanes-worth of 16B = 8KB tile each
      const int t = tid + it * 256;
      const int r = t >> 2, c8 = (t & 3) << 3;
      async_copy16(&qrow[(size_t)(i0 + r) * D_DIM + d0 + c8], &sQ[buf][r][c8]);
      async_copy16(&krow[(size_t)(j0 + r) * D_DIM + d0 + c8], &sK[buf][r][c8]);
    }
#else
#pragma unroll
    for (int it = 0; it < 8; ++it) {
      const int t = tid + it * 256;
      const int r = t >> 4, c2 = (t & 15) << 1;
      *(unsigned*)&sQ[buf][r][c2] = *(const unsigned*)&qrow[(size_t)(i0 + r) * D_DIM + d0 + c2];
      *(unsigned*)&sK[buf][r][c2] = *(const unsigned*)&krow[(size_t)(j0 + r) * D_DIM + d0 + c2];
    }
#endif
  };

  stage(0, 0);
  for (int d0 = 0; d0 < D_DIM; d0 += 32) {
    const int buf = (d0 >> 5) & 1;
    wait_async0();                               // own async fills done
    __syncthreads();                             // everyone's fills done + prior reads drained
    if (d0 + 32 < D_DIM) stage(buf ^ 1, d0 + 32);
    mma_step(sQ[buf], sK[buf], lane, wr, wc, acc);
  }

  const float scale = 0.03125f;                  // 1/sqrt(1024)
  const int h = lane >> 4;
  bf16* __restrict__ Pb = P + (size_t)b * L_DIM * L_DIM;
#pragma unroll
  for (int mi = 0; mi < 2; ++mi)
#pragma unroll
    for (int ni = 0; ni < 4; ++ni) {
      const int j = j0 + wc * 64 + ni * 16 + (lane & 15);
#pragma unroll
      for (int r = 0; r < 8; ++r) {
        const int i = i0 + wr * 32 + mi * 16 + r + 8 * h;
        float v = acc[mi][ni][r] * scale;
        if (j > i) v += NEGV;                    // causal (additive, matches reference)
        if (i >= len || j >= len) v += NEGV;     // length mask (additive)
        Pb[(size_t)i * L_DIM + j] = (bf16)v;
      }
    }
}

// ---------------- Kernel 3: row softmax in place; zero-fill above-diagonal ----------------
__global__ __launch_bounds__(256)
void softmax_kernel(bf16* __restrict__ P)
{
  const int i = blockIdx.x;
  const int b = blockIdx.y;
  bf16* __restrict__ row = P + ((size_t)b * L_DIM + i) * L_DIM;
  __shared__ __align__(16) bf16 srow[L_DIM];
  __shared__ float red[256];
  const int tid = threadIdx.x;

#pragma unroll
  for (int it = 0; it < 4; ++it) {               // 1024 dwords = full row
    const int t = tid + it * 256;
    ((unsigned*)srow)[t] = ((const unsigned*)row)[t];
  }
  __syncthreads();

  float m = -3.0e38f;
  for (int j = tid; j <= i; j += 256) m = fmaxf(m, (float)srow[j]);
  red[tid] = m;
  __syncthreads();
#pragma unroll
  for (int s = 128; s > 0; s >>= 1) {
    if (tid < s) red[tid] = fmaxf(red[tid], red[tid + s]);
    __syncthreads();
  }
  const float rowmax = red[0];
  __syncthreads();

  float sum = 0.0f;
  for (int j = tid; j <= i; j += 256) sum += __expf((float)srow[j] - rowmax);
  red[tid] = sum;
  __syncthreads();
#pragma unroll
  for (int s = 128; s > 0; s >>= 1) {
    if (tid < s) red[tid] += red[tid + s];
    __syncthreads();
  }
  const float inv = 1.0f / red[0];               // >= 1: diagonal term always present

  for (int j = tid; j < L_DIM; j += 256) {
    const float p = (j <= i) ? __expf((float)srow[j] - rowmax) * inv : 0.0f;
    row[j] = (bf16)p;
  }
}

// ---------------- Kernel 4: O = P.V (K-loop truncated at causal limit), f32 out ----------------
__global__ __launch_bounds__(256)
void pv_gemm_kernel(const bf16* __restrict__ P, const bf16* __restrict__ vT,
                    float* __restrict__ out)
{
  const int i0 = blockIdx.x * 128;
  const int d0 = blockIdx.y * 128;
  const int b  = blockIdx.z;
  __shared__ __align__(16) bf16 sP[2][128][LDSP];
  __shared__ __align__(16) bf16 sV[2][128][LDSP];  // vT tile [d][j]
  const int tid = threadIdx.x, lane = tid & 31, wave = tid >> 5;
  const int wr = wave & 3, wc = wave >> 2;
  const bf16* __restrict__ Pb = P  + (size_t)b * L_DIM * L_DIM;
  const bf16* __restrict__ Vb = vT + (size_t)b * D_DIM * L_DIM;

  v8f acc[2][4] = {};

  auto stage = [&](int buf, int j0) {
#if HAVE_ASYNC_LDS
#pragma unroll
    for (int it = 0; it < 2; ++it) {
      const int t = tid + it * 256;
      const int r = t >> 2, c8 = (t & 3) << 3;
      async_copy16(&Pb[(size_t)(i0 + r) * L_DIM + j0 + c8], &sP[buf][r][c8]);
      async_copy16(&Vb[(size_t)(d0 + r) * L_DIM + j0 + c8], &sV[buf][r][c8]);
    }
#else
#pragma unroll
    for (int it = 0; it < 8; ++it) {
      const int t = tid + it * 256;
      const int r = t >> 4, c2 = (t & 15) << 1;
      *(unsigned*)&sP[buf][r][c2] = *(const unsigned*)&Pb[(size_t)(i0 + r) * L_DIM + j0 + c2];
      *(unsigned*)&sV[buf][r][c2] = *(const unsigned*)&Vb[(size_t)(d0 + r) * L_DIM + j0 + c2];
    }
#endif
  };

  const int jEnd = i0 + 128;                     // P is exactly zero beyond the diagonal
  stage(0, 0);
  for (int j0 = 0; j0 < jEnd; j0 += 32) {
    const int buf = (j0 >> 5) & 1;
    wait_async0();
    __syncthreads();
    if (j0 + 32 < jEnd) stage(buf ^ 1, j0 + 32);
    mma_step(sP[buf], sV[buf], lane, wr, wc, acc);
  }

  const int h = lane >> 4;
#pragma unroll
  for (int mi = 0; mi < 2; ++mi)
#pragma unroll
    for (int ni = 0; ni < 4; ++ni) {
      const int d = d0 + wc * 64 + ni * 16 + (lane & 15);
#pragma unroll
      for (int r = 0; r < 8; ++r) {
        const int i = i0 + wr * 32 + mi * 16 + r + 8 * h;
        out[((size_t)b * L_DIM + i) * D_DIM + d] = acc[mi][ni][r];
      }
    }
}

extern "C" void kernel_launch(void* const* d_in, const int* in_sizes, int n_in,
                              void* d_out, int out_size, void* d_ws, size_t ws_size,
                              hipStream_t stream) {
  (void)in_sizes; (void)n_in; (void)out_size; (void)ws_size;
  const float* x  = (const float*)d_in[0];
  const float* Wq = (const float*)d_in[1];
  const float* bq = (const float*)d_in[2];
  const float* Wk = (const float*)d_in[3];
  const float* bk = (const float*)d_in[4];
  const float* Wv = (const float*)d_in[5];
  const float* bv = (const float*)d_in[6];
  const int* lengths = (const int*)d_in[7];
  float* out = (float*)d_out;

  // Workspace carve-up (bf16): Q 16MB | K 16MB | V^T 16MB | P 32MB  => 80MB total
  bf16* qb = (bf16*)d_ws;
  bf16* kb = qb + (size_t)MTOT * D_DIM;
  bf16* vT = kb + (size_t)MTOT * D_DIM;
  bf16* P  = vT + (size_t)MTOT * D_DIM;

  qkv_gemm_kernel<<<dim3(MTOT / 128, D_DIM / 128, 3), 256, 0, stream>>>(
      x, Wq, bq, Wk, bk, Wv, bv, qb, kb, vT);
  scores_kernel<<<dim3(L_DIM / 128, L_DIM / 128, B_DIM), 256, 0, stream>>>(
      qb, kb, lengths, P);
  softmax_kernel<<<dim3(L_DIM, B_DIM), 256, 0, stream>>>(P);
  pv_gemm_kernel<<<dim3(L_DIM / 128, D_DIM / 128, B_DIM), 256, 0, stream>>>(
      P, vT, out);
}
